// GraphSAGE_80676665688558
// MI455X (gfx1250) — matrix operand order repaired and checked by reading, not compiled, not run on hardware
//
#include <hip/hip_runtime.h>
#include <hip/hip_bf16.h>
#include <stdint.h>

// ---------------------------------------------------------------------------
// GraphSAGE 2-layer forward for MI455X (gfx1250, wave32, WMMA bf16)
//   h   = relu(bn( mean_agg(x) @ W1_l + x @ W1_r + b1 ))
//   out =          mean_agg(h) @ W2_l + h @ W2_r + b2
// GEMMs use v_wmma_f32_16x16x32_bf16 (fp32 accumulate). Aggregation uses
// f32 global atomics (accumulator lives in the 192MB L2).
// ---------------------------------------------------------------------------

typedef __attribute__((ext_vector_type(16))) __bf16 v16bf;
typedef __attribute__((ext_vector_type(8)))  float  v8f;

#define N_CH   128
#define BN_EPS 1e-5f

__device__ __forceinline__ uint32_t pack2bf(float a, float b) {
  __bf16 x = (__bf16)a, y = (__bf16)b;
  uint16_t ux, uy;
  __builtin_memcpy(&ux, &x, 2);
  __builtin_memcpy(&uy, &y, 2);
  return (uint32_t)ux | ((uint32_t)uy << 16);
}

// ---- zero scratch (agg1, agg2, cnt) ---------------------------------------
__global__ void zero_f4(float4* __restrict__ p, long n4) {
  long i = (long)blockIdx.x * blockDim.x + threadIdx.x;
  if (i < n4) p[i] = make_float4(0.f, 0.f, 0.f, 0.f);
}

// ---- in-degree ------------------------------------------------------------
__global__ void degree_k(const int* __restrict__ dst, float* __restrict__ cnt, int E) {
  int e = blockIdx.x * blockDim.x + threadIdx.x;
  if (e < E) atomicAdd(&cnt[dst[e]], 1.0f);
}

// ---- edge scatter-add: agg[dst] += feat[src]  (128 ch, 4 ch per thread) ---
__global__ void scatter_k(const float* __restrict__ feat, const int* __restrict__ src,
                          const int* __restrict__ dst, float* __restrict__ agg, int E) {
  long gid = (long)blockIdx.x * blockDim.x + threadIdx.x;
  int e = (int)(gid >> 5);
  if (e >= E) return;
  int c = ((int)gid & 31) << 2;
  float4 v = *(const float4*)(feat + (long)src[e] * N_CH + c);
  float* o = agg + (long)dst[e] * N_CH + c;
  atomicAdd(o + 0, v.x);
  atomicAdd(o + 1, v.y);
  atomicAdd(o + 2, v.z);
  atomicAdd(o + 3, v.w);
}

// ---- weight prep: [W_l;W_r] (K=256) -> bf16 WMMA B-fragments --------------
// Fragment f = kt*NTILES+nt holds the 32x16 B tile in per-lane order:
//   dword index = lane*8 + v ; lane<16: K = k0+2v,2v+1 col=lane
//                              lane>=16: K = k0+16+2v,+1 col=lane-16
__global__ void prep_weights(const float* __restrict__ W1l, const float* __restrict__ W1r,
                             const float* __restrict__ W2l, const float* __restrict__ W2r,
                             uint32_t* __restrict__ frag1, uint32_t* __restrict__ frag2) {
  int tid = blockIdx.x * blockDim.x + threadIdx.x;
  if (tid < 64 * 256) {                      // layer 1: 8 kt x 8 nt fragments
    int f = tid >> 8, w = tid & 255;
    int lane = w >> 3, v = w & 7;
    int kt = f >> 3, nt = f & 7;
    int col = nt * 16 + (lane & 15);
    int k = kt * 32 + ((lane >> 4) ? 16 : 0) + 2 * v;
    float x0 = (k < 128)     ? W1l[k * 128 + col]           : W1r[(k - 128) * 128 + col];
    float x1 = (k + 1 < 128) ? W1l[(k + 1) * 128 + col]     : W1r[(k - 127) * 128 + col];
    frag1[tid] = pack2bf(x0, x1);
  } else if (tid < 64 * 256 + 32 * 256) {    // layer 2: 8 kt x 4 nt fragments
    int t = tid - 64 * 256;
    int f = t >> 8, w = t & 255;
    int lane = w >> 3, v = w & 7;
    int kt = f >> 2, nt = f & 3;
    int col = nt * 16 + (lane & 15);
    int k = kt * 32 + ((lane >> 4) ? 16 : 0) + 2 * v;
    float x0 = (k < 128)     ? W2l[k * 64 + col]            : W2r[(k - 128) * 64 + col];
    float x1 = (k + 1 < 128) ? W2l[(k + 1) * 64 + col]      : W2r[(k - 127) * 64 + col];
    frag2[t] = pack2bf(x0, x1);
  }
}

// ---- fused SAGE GEMM: out = (agg/deg)@W_l + self@W_r + b [+ BN + ReLU] ----
// One wave per 16x16 output tile, K=256 (concatenated [agg | self]),
// 8 steps of v_wmma_f32_16x16x32_bf16.
template <int NTILES, bool LAYER1>
__global__ void sage_gemm(const float* __restrict__ aggA, const float* __restrict__ selfX,
                          const float* __restrict__ cnt, const uint32_t* __restrict__ frag,
                          const float* __restrict__ bias,
                          const float* __restrict__ g, const float* __restrict__ be,
                          const float* __restrict__ mu, const float* __restrict__ va,
                          float* __restrict__ out, int mtiles) {
  const int wave = threadIdx.x >> 5;
  const int lane = threadIdx.x & 31;
  const int mTile = blockIdx.x * 8 + wave;
  if (mTile >= mtiles) return;                 // whole-wave exit: EXEC stays all-1s
  const int nt = blockIdx.y;
  const int r = lane & 15;
  const int half = lane >> 4;
  const long row = (long)mTile * 16 + r;       // A-matrix: both lane halves hold row r
  const float inv = 1.0f / fmaxf(cnt[row], 1.0f);

  v8f acc = {};
#pragma unroll
  for (int kt = 0; kt < 8; ++kt) {
    const int k0 = kt * 32;
    const float* sp;
    float s;
    if (k0 < N_CH) { sp = aggA + row * N_CH + k0;          s = inv;  }
    else           { sp = selfX + row * N_CH + (k0 - N_CH); s = 1.0f; }
    // A k order used here (matches B frags): lane<16 -> K {0..15}, lane>=16 -> K {16..31}
    const float* q0 = sp + half * 16;
    float4 f0 = ((const float4*)q0)[0];
    float4 f1 = ((const float4*)q0)[1];
    float4 f2 = ((const float4*)(q0 + 8))[0];
    float4 f3 = ((const float4*)(q0 + 8))[1];
    v16bf a;
    a[0]  = (__bf16)(f0.x * s); a[1]  = (__bf16)(f0.y * s);
    a[2]  = (__bf16)(f0.z * s); a[3]  = (__bf16)(f0.w * s);
    a[4]  = (__bf16)(f1.x * s); a[5]  = (__bf16)(f1.y * s);
    a[6]  = (__bf16)(f1.z * s); a[7]  = (__bf16)(f1.w * s);
    a[8]  = (__bf16)(f2.x * s); a[9]  = (__bf16)(f2.y * s);
    a[10] = (__bf16)(f2.z * s); a[11] = (__bf16)(f2.w * s);
    a[12] = (__bf16)(f3.x * s); a[13] = (__bf16)(f3.y * s);
    a[14] = (__bf16)(f3.z * s); a[15] = (__bf16)(f3.w * s);
    // B fragment: 32 contiguous bytes per lane (pre-swizzled by prep_weights)
    const uint32_t* bp = frag + (((kt * NTILES) + nt) << 8) + (lane << 3);
    union { uint4 u[2]; v16bf b; } ub;
    ub.u[0] = ((const uint4*)bp)[0];
    ub.u[1] = ((const uint4*)bp)[1];
    acc = __builtin_amdgcn_wmma_f32_16x16x32_bf16(
        /*neg_a=*/false, a, /*neg_b=*/false, ub.b,
        /*c_mod=*/(short)0, acc, /*reuse_a=*/false, /*reuse_b=*/false);
  }

  // C/D layout: VGPR i -> row (i + half*8), lane -> col (lane&15)
  const int col = nt * 16 + r;
  const float bb = bias[col];
  float sc = 1.0f, muc = 0.0f, bec = 0.0f;
  if (LAYER1) {
    sc  = g[col] * rsqrtf(va[col] + BN_EPS);
    muc = mu[col];
    bec = be[col];
  }
  const int ldo = NTILES * 16;
#pragma unroll
  for (int i = 0; i < 8; ++i) {
    long orow = (long)mTile * 16 + half * 8 + i;
    float v = acc[i] + bb;
    if (LAYER1) {
      v = (v - muc) * sc + bec;
      v = fmaxf(v, 0.0f);
    }
    out[orow * ldo + col] = v;
  }
}

// ---------------------------------------------------------------------------
extern "C" void kernel_launch(void* const* d_in, const int* in_sizes, int n_in,
                              void* d_out, int out_size, void* d_ws, size_t ws_size,
                              hipStream_t stream) {
  const float* x     = (const float*)d_in[0];
  const int*   edges = (const int*)d_in[1];      // [2,E]: row0=src, row1=dst
  const float* W1l   = (const float*)d_in[2];
  const float* W1r   = (const float*)d_in[3];
  const float* b1    = (const float*)d_in[4];
  const float* gam   = (const float*)d_in[5];
  const float* bet   = (const float*)d_in[6];
  const float* mean  = (const float*)d_in[7];
  const float* var   = (const float*)d_in[8];
  const float* W2l   = (const float*)d_in[9];
  const float* W2r   = (const float*)d_in[10];
  const float* b2    = (const float*)d_in[11];

  const int N = in_sizes[0] / N_CH;   // 50000 (divisible by 16)
  const int E = in_sizes[1] / 2;      // 600000
  const int* srcIdx = edges;
  const int* dstIdx = edges + E;

  // workspace layout: [agg1][agg2][cnt][h][frag1][frag2]
  const size_t nc = (size_t)N * N_CH;
  float*    agg1  = (float*)d_ws;
  float*    agg2  = agg1 + nc;
  float*    cnt   = agg2 + nc;
  float*    h     = cnt + N;
  uint32_t* frag1 = (uint32_t*)(h + nc);
  uint32_t* frag2 = frag1 + 64 * 256;

  const int mtiles  = (N + 15) / 16;          // 3125
  const int blocksM = (mtiles + 7) / 8;       // 8 waves (16x16 tiles) per block

  // 1) zero accumulators (agg1, agg2, cnt are contiguous)
  long zn4 = (long)(2 * nc + (size_t)N) / 4;
  zero_f4<<<(int)((zn4 + 255) / 256), 256, 0, stream>>>((float4*)d_ws, zn4);

  // 2) bf16 weight fragments for both layers
  prep_weights<<<96, 256, 0, stream>>>(W1l, W1r, W2l, W2r, frag1, frag2);

  // 3) in-degree (shared by both layers)
  degree_k<<<(E + 255) / 256, 256, 0, stream>>>(dstIdx, cnt, E);

  // 4) layer-1 aggregation: agg1[dst] += x[src]
  scatter_k<<<(int)(((long)E * 32 + 255) / 256), 256, 0, stream>>>(x, srcIdx, dstIdx, agg1, E);

  // 5) h = relu(bn((agg1/deg)@W1_l + x@W1_r + b1))
  {
    dim3 grid(blocksM, 8);
    sage_gemm<8, true><<<grid, 256, 0, stream>>>(agg1, x, cnt, frag1, b1,
                                                 gam, bet, mean, var, h, mtiles);
  }

  // 6) layer-2 aggregation: agg2[dst] += h[src]
  scatter_k<<<(int)(((long)E * 32 + 255) / 256), 256, 0, stream>>>(h, srcIdx, dstIdx, agg2, E);

  // 7) out = (agg2/deg)@W2_l + h@W2_r + b2
  {
    dim3 grid(blocksM, 4);
    sage_gemm<4, false><<<grid, 256, 0, stream>>>(agg2, h, cnt, frag2, b2,
                                                  nullptr, nullptr, nullptr, nullptr,
                                                  (float*)d_out, mtiles);
  }
}
